// CrossAttention_7258494730671
// MI455X (gfx1250) — matrix operand order repaired and checked by reading, not compiled, not run on hardware
//
#include <hip/hip_runtime.h>

// ---- problem constants ----
#define Bc   32
#define NQc  1024
#define NKc  2048
#define Dc   256     // == DQ == DK

typedef __attribute__((ext_vector_type(16))) __bf16    v16bf;
typedef __attribute__((ext_vector_type(2)))  __bf16    v2bf;
typedef __attribute__((ext_vector_type(8)))  float     v8f;
typedef __attribute__((ext_vector_type(8)))  unsigned  v8u;
typedef __attribute__((ext_vector_type(4)))  unsigned  v4u;
typedef __attribute__((ext_vector_type(4)))  float     v4f;
typedef __attribute__((ext_vector_type(4)))  int       v4i;

#ifdef __has_builtin
#if __has_builtin(__builtin_amdgcn_cvt_pk_bf16_f32)
#define HAVE_CVT_PK 1
#endif
#if __has_builtin(__builtin_amdgcn_global_store_async_from_lds_b128)
#define HAVE_ASYNC_STORE 1
#endif
#endif
#ifndef HAVE_CVT_PK
#define HAVE_CVT_PK 0
#endif
#ifndef HAVE_ASYNC_STORE
#define HAVE_ASYNC_STORE 0
#endif

#if HAVE_ASYNC_STORE
// builtin signature (from clang diagnostic): (v4i* [global], v4i* [lds], imm, imm)
typedef v4i __attribute__((address_space(1)))* gv4p;
typedef v4i __attribute__((address_space(3)))* lv4p;
#endif

static __device__ inline unsigned pack_bf16(float lo, float hi) {
#if HAVE_CVT_PK
  v2bf r = __builtin_amdgcn_cvt_pk_bf16_f32(lo, hi);
  return __builtin_bit_cast(unsigned, r);
#else
  unsigned ulo = __builtin_bit_cast(unsigned, lo);
  unsigned uhi = __builtin_bit_cast(unsigned, hi);
  return (ulo >> 16) | (uhi & 0xffff0000u);
#endif
}
static __device__ inline unsigned short to_bf16(float v) {
  return (unsigned short)(__builtin_bit_cast(unsigned, v) >> 16);
}

// A fragment (16x32 bf16, MxK) from bf16 memory/LDS: lane (r=lane&15, h=lane>>4)
// loads 8 bf16 at p0 = row + h*16 (K=h*16+0..7) and 8 at p1 = p0+8 (K=h*16+8..15)
static __device__ inline v16bf load_A_bf16(const unsigned short* p0,
                                           const unsigned short* p1) {
  v4u lo = *(const v4u*)p0;
  v4u hi = *(const v4u*)p1;
  v8u u = {lo[0], lo[1], lo[2], lo[3], hi[0], hi[1], hi[2], hi[3]};
  return __builtin_bit_cast(v16bf, u);
}
// B fragment (32x16 bf16, KxN): lane (c=lane&15, h=lane>>4) loads 16 contiguous
// bf16 at Brow(c) + h*16  (K = h*16 + 0..15)
static __device__ inline v16bf load_B_bf16(const unsigned short* p) {
  return __builtin_bit_cast(v16bf, *(const v8u*)p);
}

static __device__ inline v8f wmma_bf16(v16bf a, v16bf b, v8f c) {
  return __builtin_amdgcn_wmma_f32_16x16x32_bf16(false, a, false, b,
                                                 (short)0, c, false, false);
}

// ------------------------------------------------------------------
// One-time fp32 -> bf16 conversion of a weight matrix (row-major kept).
// ------------------------------------------------------------------
__global__ __launch_bounds__(256) void cvtw_kernel(
    const float* __restrict__ in, unsigned short* __restrict__ out, int npairs) {
  int i = blockIdx.x * blockDim.x + threadIdx.x;
  if (i < npairs) ((unsigned*)out)[i] = pack_bf16(in[2 * i], in[2 * i + 1]);
}

// ------------------------------------------------------------------
// Projection: out[row, n] = (sum_k in[row,k] * W[n,k] + bias[n]) * scale
// One block per 16-row block: stage A as bf16 in LDS once, 8 waves cover all
// 16 column tiles (2 per wave). W pre-converted to bf16 (row-major).
// transposed=1 stores out as (batch, Dc, nk) i.e. v^T.
// ------------------------------------------------------------------
__global__ __launch_bounds__(256) void proj_kernel(
    const float* __restrict__ in, const unsigned short* __restrict__ Wbf,
    const float* __restrict__ bias, float scale,
    unsigned short* __restrict__ out, int transposed, int nk) {
  __shared__ unsigned short aS[16 * Dc];   // 8 KB
  int tid = threadIdx.x;
  int mr = blockIdx.x * 16;

  { // cooperative fp32->bf16 staging of the 16x256 A row-block
    int r2 = tid >> 4;
    int g2 = (tid & 15) * 16;
    const float* src = in + (long)(mr + r2) * Dc + g2;
    unsigned* dst = (unsigned*)(aS + r2 * Dc + g2);
#pragma unroll
    for (int i = 0; i < 8; ++i) dst[i] = pack_bf16(src[2 * i], src[2 * i + 1]);
  }
  __syncthreads();

  int w = tid >> 5, lane = tid & 31;
  int r = lane & 15, h = lane >> 4, cl = lane & 15;
  int n0 = w * 32;

  const unsigned short* arow = aS + r * Dc;
  const unsigned short* b0r = Wbf + (long)(n0 + cl) * Dc;
  const unsigned short* b1r = Wbf + (long)(n0 + 16 + cl) * Dc;

  v8f c0 = {}, c1 = {};
#pragma unroll
  for (int kk = 0; kk < Dc; kk += 32) {
    v16bf a = load_A_bf16(arow + kk + h * 16, arow + kk + h * 16 + 8);
    c0 = wmma_bf16(a, load_B_bf16(b0r + kk + h * 16), c0);
    c1 = wmma_bf16(a, load_B_bf16(b1r + kk + h * 16), c1);
  }

  float bn0 = bias[n0 + cl];
  float bn1 = bias[n0 + 16 + cl];
  if (!transposed) {
#pragma unroll
    for (int j = 0; j < 8; ++j) {
      int moff = j + h * 8;
      out[(long)(mr + moff) * Dc + n0 + cl]      = to_bf16((c0[j] + bn0) * scale);
      out[(long)(mr + moff) * Dc + n0 + 16 + cl] = to_bf16((c1[j] + bn1) * scale);
    }
  } else {
    int bb = mr / nk;               // tile never crosses a batch (nk % 16 == 0)
    int nkbase = mr - bb * nk;
#pragma unroll
    for (int j = 0; j < 8; ++j) {
      int moff = j + h * 8;
      out[((long)bb * Dc + n0 + cl) * nk + nkbase + moff]      = to_bf16((c0[j] + bn0) * scale);
      out[((long)bb * Dc + n0 + 16 + cl) * nk + nkbase + moff] = to_bf16((c1[j] + bn1) * scale);
    }
  }
}

// ------------------------------------------------------------------
// Attention: one block (8 waves, 256 thr) per (batch, 16-row q tile).
// LDS: 16x2048 fp32 logits (128 KB) + 16x2048 bf16 weights (64 KB).
// ------------------------------------------------------------------
__global__ __launch_bounds__(256) void attn_kernel(
    const unsigned short* __restrict__ qhat,   // (B*NQ, 256) bf16 (scale folded)
    const unsigned short* __restrict__ khat,   // (B*NK, 256) bf16
    const unsigned short* __restrict__ vT,     // (B, 256, NK) bf16
    const int* __restrict__ qmask, const int* __restrict__ kmask,
    const float* __restrict__ bias2,           // (B, NQ, NK) fp32
    float* __restrict__ outp,                  // (B, NQ, 256) fp32
    float* __restrict__ weights) {             // (B, NQ, NK) fp32
  extern __shared__ float smem[];
  float* lg = smem;                                       // 16*2048 fp32
  unsigned short* wbf = (unsigned short*)(lg + 16 * NKc); // 16*2048 bf16
  float* red    = (float*)(wbf + 16 * NKc);               // 256
  float* rowmax = red + 256;                              // 16
  float* rowinv = rowmax + 16;                            // 16

  int b   = blockIdx.x >> 6;            // / (NQ/16)
  int mq0 = (blockIdx.x & 63) << 4;
  int tid = threadIdx.x;
  int w    = tid >> 5;
  int lane = tid & 31;
  int r    = lane & 15;
  int h    = lane >> 4;
  int cl   = lane & 15;

  // ---------------- Phase 1: logits tile -> LDS ----------------
  v16bf a_all[8];
  const unsigned short* qrow = qhat + (long)(b * NQc + mq0 + r) * Dc;
#pragma unroll
  for (int kv = 0; kv < 8; ++kv)
    a_all[kv] = load_A_bf16(qrow + kv * 32 + h * 16, qrow + kv * 32 + h * 16 + 8);

  int qm[8];
#pragma unroll
  for (int j = 0; j < 8; ++j)
    qm[j] = qmask[b * NQc + mq0 + j + h * 8];

  const float* bbase = bias2 + ((long)b * NQc + mq0) * NKc;

  for (int tt = 0; tt < 16; ++tt) {
    int n0 = w * 256 + tt * 16;
    int kmv = kmask[b * NKc + n0 + cl];
    const unsigned short* krow = khat + (long)(b * NKc + n0 + cl) * Dc;
    v8f c = {};
#pragma unroll
    for (int kv = 0; kv < 8; ++kv)
      c = wmma_bf16(a_all[kv], load_B_bf16(krow + kv * 32 + h * 16), c);
    const float* bput = bbase + n0 + cl;
#pragma unroll
    for (int j = 0; j < 8; ++j) {
      int moff = j + h * 8;
      float v = c[j] + bput[(long)moff * NKc];
      if ((qm[j] == 0) | (kmv == 0)) v = -1e9f;
      lg[moff * NKc + n0 + cl] = v;
    }
  }
  __syncthreads();

  // ------- Phase 2: softmax in LDS; bf16 copy for phase 3; stream weights ----
  {
    int rr = tid >> 4;                 // row 0..15
    int g  = tid & 15;                 // 16 threads per row
    float* myrow = lg + rr * NKc;
    float lmax = -3.4e38f;
#pragma unroll 4
    for (int c = g * 128; c < g * 128 + 128; ++c) lmax = fmaxf(lmax, myrow[c]);
    red[tid] = lmax;
    __syncthreads();
    if (g == 0) {
      float m = red[tid];
#pragma unroll
      for (int i = 1; i < 16; ++i) m = fmaxf(m, red[tid + i]);
      rowmax[rr] = m;
    }
    __syncthreads();
    float m = rowmax[rr];
    float lsum = 0.f;
#pragma unroll 4
    for (int c = g * 128; c < g * 128 + 128; ++c) {
      float e = __expf(myrow[c] - m);
      myrow[c] = e;
      lsum += e;
    }
    red[tid] = lsum;
    __syncthreads();
    if (g == 0) {
      float s = 0.f;
#pragma unroll
      for (int i = 0; i < 16; ++i) s += red[tid + i];
      rowinv[rr] = 1.0f / s;
    }
    __syncthreads();
    float inv = rowinv[rr];
    v4f* row4 = (v4f*)myrow;
    unsigned* wb32 = (unsigned*)(wbf + rr * NKc);
    float* wrow = weights + ((long)b * NQc + mq0 + rr) * NKc;
#pragma unroll 4
    for (int i = 0; i < 32; ++i) {
      int idx = i * 16 + g;
      v4f x = row4[idx];
      x *= inv;
      row4[idx] = x;                         // normalized fp32 (async-store src)
      wb32[idx * 2]     = pack_bf16(x[0], x[1]);   // bf16 copy for phase 3
      wb32[idx * 2 + 1] = pack_bf16(x[2], x[3]);
#if !HAVE_ASYNC_STORE
      *((v4f*)wrow + idx) = x;
#endif
    }
#if HAVE_ASYNC_STORE
    __syncthreads();   // ensure ds stores retired before async engine reads LDS
#pragma unroll 4
    for (int i = 0; i < 32; ++i) {
      int idx = i * 16 + g;
      __builtin_amdgcn_global_store_async_from_lds_b128(
          (gv4p)(wrow + idx * 4), (lv4p)(myrow + idx * 4), 0, 0);
    }
#endif
  }
  __syncthreads();

  // -------- Phase 3: out = weights(16xNK) x vT -> 16x32 per wave (bf16 A) ----
  v8f c0 = {}, c1 = {};
  int d0 = w * 32;
  const unsigned short* vb = vT + (long)b * Dc * NKc;
  const unsigned short* wrowb = wbf + r * NKc;
  for (int kk = 0; kk < NKc; kk += 32) {
    v16bf a  = load_A_bf16(wrowb + kk + h * 16, wrowb + kk + h * 16 + 8);
    v16bf b0 = load_B_bf16(vb + (long)(d0 + cl) * NKc + kk + h * 16);
    v16bf b1 = load_B_bf16(vb + (long)(d0 + 16 + cl) * NKc + kk + h * 16);
    c0 = wmma_bf16(a, b0, c0);
    c1 = wmma_bf16(a, b1, c1);
  }
  float* orow = outp + ((long)b * NQc + mq0) * Dc;
#pragma unroll
  for (int j = 0; j < 8; ++j) {
    int moff = j + h * 8;
    orow[(long)moff * Dc + d0 + cl]      = c0[j];
    orow[(long)moff * Dc + d0 + 16 + cl] = c1[j];
  }
}

extern "C" void kernel_launch(void* const* d_in, const int* in_sizes, int n_in,
                              void* d_out, int out_size, void* d_ws, size_t ws_size,
                              hipStream_t stream) {
  (void)in_sizes; (void)n_in; (void)out_size; (void)ws_size;
  const float* q_in  = (const float*)d_in[0];
  const float* k_in  = (const float*)d_in[1];
  const float* v_in  = (const float*)d_in[2];
  const int*   qmask = (const int*)d_in[3];
  const int*   kmask = (const int*)d_in[4];
  const float* bias2 = (const float*)d_in[5];
  const float* Wq = (const float*)d_in[6];
  const float* bq = (const float*)d_in[7];
  const float* Wk = (const float*)d_in[8];
  const float* bk = (const float*)d_in[9];
  const float* Wv = (const float*)d_in[10];
  const float* bv = (const float*)d_in[11];

  unsigned short* qhat = (unsigned short*)d_ws;                 // 32768*256 bf16
  unsigned short* khat = qhat + (size_t)Bc * NQc * Dc;          // 65536*256 bf16
  unsigned short* vT   = khat + (size_t)Bc * NKc * Dc;          // 65536*256 bf16
  unsigned short* Wqb  = vT + (size_t)Bc * NKc * Dc;            // 256*256 bf16
  unsigned short* Wkb  = Wqb + (size_t)Dc * Dc;
  unsigned short* Wvb  = Wkb + (size_t)Dc * Dc;

  float* outp    = (float*)d_out;
  float* weights = outp + (size_t)Bc * NQc * Dc;

  const int qrows = Bc * NQc;   // 32768
  const int krows = Bc * NKc;   // 65536
  const int wpairs = Dc * Dc / 2;  // 32768

  cvtw_kernel<<<wpairs / 256, 256, 0, stream>>>(Wq, Wqb, wpairs);
  cvtw_kernel<<<wpairs / 256, 256, 0, stream>>>(Wk, Wkb, wpairs);
  cvtw_kernel<<<wpairs / 256, 256, 0, stream>>>(Wv, Wvb, wpairs);

  // scale 1/sqrt(256) folded into q-hat
  proj_kernel<<<qrows / 16, 256, 0, stream>>>(q_in, Wqb, bq, 1.0f / 16.0f, qhat, 0, 0);
  proj_kernel<<<krows / 16, 256, 0, stream>>>(k_in, Wkb, bk, 1.0f, khat, 0, 0);
  proj_kernel<<<krows / 16, 256, 0, stream>>>(v_in, Wvb, bv, 1.0f, vT, 1, NKc);

  size_t smem = (size_t)(16 * NKc * 4 + 16 * NKc * 2 + (256 + 32) * 4);  // ~193 KB
  attn_kernel<<<Bc * (NQc / 16), 256, smem, stream>>>(
      qhat, khat, vT, qmask, kmask, bias2, outp, weights);
}
